// LinearAffine8bit_11355893531299
// MI455X (gfx1250) — compile-verified, matrix operand order checked by reading, not run on hardware
//
#include <hip/hip_runtime.h>

typedef _Float16 v16h __attribute__((ext_vector_type(16)));
typedef _Float16 v8h  __attribute__((ext_vector_type(8)));
typedef _Float16 v4h  __attribute__((ext_vector_type(4)));
typedef float    v8f  __attribute__((ext_vector_type(8)));

#define M_TOTAL 8192   // B*S = 4*2048
#define N_TOTAL 4096   // OUT
#define K_TOTAL 4096   // IN
#define BM 128
#define BN 128
#define BK 32
#define PITCH 40                 // BK + 8 halves of padding (80B row pitch, 16B aligned)
#define NKITER (K_TOTAL / BK)    // 128

__global__ __launch_bounds__(256)
void qlinear_wmma_f16(const float* __restrict__ X,     // [M_TOTAL, K_TOTAL] fp32
                      const int*   __restrict__ Q,     // [N_TOTAL, K_TOTAL] int32 (0..255)
                      const float* __restrict__ scale_p,
                      const float* __restrict__ zp_p,
                      const float* __restrict__ bias,  // [N_TOTAL]
                      float*       __restrict__ Out)   // [M_TOTAL, N_TOTAL]
{
    __shared__ _Float16 sA[2 * BM * PITCH];  // x tile, f16
    __shared__ _Float16 sB[2 * BN * PITCH];  // (q - zp) tile, f16

    const int tid  = threadIdx.x;
    const int lane = tid & 31;
    const int wave = tid >> 5;
    const int wm   = wave & 1;   // 0..1 -> 64 rows each
    const int wn   = wave >> 1;  // 0..3 -> 32 cols each

    const int blk_n = blockIdx.x;  // 0..31
    const int blk_m = blockIdx.y;  // 0..63

    const float scale = scale_p[0];
    const float zp    = zp_p[0];

    // ---- global->reg staging: 4 x float4 of X, 4 x int4 of Q per thread ----
    float4 xr[4];
    int4   qr[4];

    auto gload = [&](int kt) {
        const float4* xg = (const float4*)X;
        const int4*   qg = (const int4*)Q;
        const int kcol0 = kt * (BK / 4);  // in 4-element units
#pragma unroll
        for (int q = 0; q < 4; ++q) {
            int idx = tid + q * 256;      // 0..1023 float4 slots of the 128x32 tile
            int row = idx >> 3;           // 0..127
            int c4  = idx & 7;            // 0..7
            xr[q] = xg[(size_t)(blk_m * BM + row) * (K_TOTAL / 4) + kcol0 + c4];
            qr[q] = qg[(size_t)(blk_n * BN + row) * (K_TOTAL / 4) + kcol0 + c4];
        }
    };

    auto sstore = [&](int buf) {
        _Float16* A = sA + buf * (BM * PITCH);
        _Float16* B = sB + buf * (BN * PITCH);
#pragma unroll
        for (int q = 0; q < 4; ++q) {
            int idx = tid + q * 256;
            int row = idx >> 3;
            int c4  = idx & 7;
            v4h ha;
            ha[0] = (_Float16)xr[q].x; ha[1] = (_Float16)xr[q].y;
            ha[2] = (_Float16)xr[q].z; ha[3] = (_Float16)xr[q].w;
            *(v4h*)(A + row * PITCH + c4 * 4) = ha;
            v4h hb;
            hb[0] = (_Float16)((float)qr[q].x - zp);
            hb[1] = (_Float16)((float)qr[q].y - zp);
            hb[2] = (_Float16)((float)qr[q].z - zp);
            hb[3] = (_Float16)((float)qr[q].w - zp);
            *(v4h*)(B + row * PITCH + c4 * 4) = hb;
        }
    };

    // ---- WMMA fragment loader (A 16x32 / B 32x16 mirrored layout, ISA 7.12.2) ----
    const int fr = lane & 15;             // row (A: m) or col (B: n) within 16
    const int kb = (lane >> 4) << 3;      // kbase: 0 or 8
    auto load_frag = [&](const _Float16* tile, int row0) -> v16h {
        const _Float16* p = tile + (row0 + fr) * PITCH + kb;
        v8h lo = *(const v8h*)(p);        // K = kb .. kb+7
        v8h hi = *(const v8h*)(p + 16);   // K = 16+kb .. 16+kb+7
        v16h r;
#pragma unroll
        for (int i = 0; i < 8; ++i) { r[i] = lo[i]; r[8 + i] = hi[i]; }
        return r;
    };

    v8f acc[4][2];
#pragma unroll
    for (int i = 0; i < 4; ++i)
#pragma unroll
        for (int j = 0; j < 2; ++j)
#pragma unroll
            for (int r = 0; r < 8; ++r) acc[i][j][r] = 0.0f;

    gload(0);
    sstore(0);
    __syncthreads();

    for (int kt = 0; kt < NKITER; ++kt) {
        const int buf = kt & 1;

        // Issue next tile's global loads first, and pin them there so the
        // ~12 ds_load + 8 wmma compute phase below covers their latency.
        if (kt + 1 < NKITER) gload(kt + 1);
        __builtin_amdgcn_sched_barrier(0);

        const _Float16* A = sA + buf * (BM * PITCH);
        const _Float16* B = sB + buf * (BN * PITCH);

        v16h af[4], bf[2];
#pragma unroll
        for (int i = 0; i < 4; ++i) af[i] = load_frag(A, wm * 64 + i * 16);
#pragma unroll
        for (int j = 0; j < 2; ++j) bf[j] = load_frag(B, wn * 32 + j * 16);

#pragma unroll
        for (int i = 0; i < 4; ++i)
#pragma unroll
            for (int j = 0; j < 2; ++j)
                acc[i][j] = __builtin_amdgcn_wmma_f32_16x16x32_f16(
                    /*neg_a=*/false, af[i], /*neg_b=*/false, bf[j],
                    /*c_mod=*/(short)0, acc[i][j],
                    /*reuse_a=*/false, /*reuse_b=*/false);

        __builtin_amdgcn_sched_barrier(0);
        if (kt + 1 < NKITER) sstore((kt + 1) & 1);
        __syncthreads();
    }

    // ---- epilogue: out = scale*acc + bias ; C layout: m = r + 8*(lane>>4), n = lane&15 ----
    // Non-temporal stores: keep x (128MB) + qweight (64MB) resident in the 192MB L2,
    // don't let the write-once 128MB output evict them.
#pragma unroll
    for (int i = 0; i < 4; ++i) {
        const int mbase = blk_m * BM + wm * 64 + i * 16 + ((lane >> 4) << 3);
#pragma unroll
        for (int j = 0; j < 2; ++j) {
            const int n  = blk_n * BN + wn * 32 + j * 16 + (lane & 15);
            const float bv = bias[n];
#pragma unroll
            for (int r = 0; r < 8; ++r) {
                const int m = mbase + r;
                __builtin_nontemporal_store(scale * acc[i][j][r] + bv,
                                            &Out[(size_t)m * N_TOTAL + n]);
            }
        }
    }
}

extern "C" void kernel_launch(void* const* d_in, const int* in_sizes, int n_in,
                              void* d_out, int out_size, void* d_ws, size_t ws_size,
                              hipStream_t stream) {
    const float* x     = (const float*)d_in[0];
    const int*   qw    = (const int*)d_in[1];
    const float* scale = (const float*)d_in[2];
    const float* zp    = (const float*)d_in[3];
    const float* bias  = (const float*)d_in[4];
    float* out = (float*)d_out;

    dim3 grid(N_TOTAL / BN, M_TOTAL / BM);  // 32 x 64
    qlinear_wmma_f16<<<grid, 256, 0, stream>>>(x, qw, scale, zp, bias, out);
}
